// MultiHeadAttention_46523085751035
// MI455X (gfx1250) — matrix operand order repaired and checked by reading, not compiled, not run on hardware
//
#include <hip/hip_runtime.h>
#include <hip/hip_bf16.h>

// ---------------------------------------------------------------------------
// MHA forward for MI455X (gfx1250, wave32, WMMA f16 -> f32 accumulate)
// B=2, S=2048, E=1024, H=16, HS=64.  Reshape (B,S,E)->(B,H,S,HS) is a pure
// reinterpretation: each head is a contiguous S x HS block of the flat buffer.
// ---------------------------------------------------------------------------

#define BB 2
#define SS 2048
#define EE 1024
#define HH 16
#define HSZ 64
#define MM (BB * SS)          // 4096 rows in all projection GEMMs

typedef __attribute__((ext_vector_type(16))) _Float16 v16h;
typedef __attribute__((ext_vector_type(8)))  float    v8f;

union H16 {
    uint4    u2[2];
    _Float16 h[16];
    v16h     v;
};

__device__ __forceinline__ v16h ld2x16(const _Float16* p0, const _Float16* p1) {
    H16 t;
    t.u2[0] = *(const uint4*)p0;
    t.u2[1] = *(const uint4*)p1;
    return t.v;
}

__device__ __forceinline__ v8f wmma32(v16h a, v16h b, v8f c) {
    // D = A(16x32 f16) * B(32x16 f16) + C(16x16 f32)
    return __builtin_amdgcn_wmma_f32_16x16x32_f16(
        /*neg_a=*/false, a, /*neg_b=*/false, b,
        /*c_mod=*/(short)0, c, /*reuse_a=*/false, /*reuse_b=*/false);
}

// ---------------------------------------------------------------------------
// fp32 -> f16 conversion (n multiple of 1024)
// ---------------------------------------------------------------------------
__global__ void cvt_f32_to_f16(const float* __restrict__ s,
                               _Float16* __restrict__ d, int n) {
    int i = (blockIdx.x * 256 + threadIdx.x) * 4;
    if (i + 3 < n) {
        float4 v = *(const float4*)(s + i);
        d[i + 0] = (_Float16)v.x;
        d[i + 1] = (_Float16)v.y;
        d[i + 2] = (_Float16)v.z;
        d[i + 3] = (_Float16)v.w;
    }
}

// ---------------------------------------------------------------------------
// Tiled f16 GEMM:  C(M x N) = A(M x K, row-major f16) * Bw(K x N, row-major f16)
// + bias, * scale.  BM=128, BN=128, BK=32.  256 threads = 8 waves; each wave
// owns a 32x64 tile (2x4 WMMA accumulators).
// ---------------------------------------------------------------------------
__global__ void gemm_f16_kernel(const _Float16* __restrict__ A,
                                const _Float16* __restrict__ Bw,
                                const float* __restrict__ bias,
                                _Float16* __restrict__ outh,
                                float* __restrict__ outf,
                                int Mdim, int Ndim, int Kdim, float scale) {
    __shared__ _Float16 As[128 * 40];   // [row][k], padded stride 40 halves
    __shared__ _Float16 Bt[128 * 40];   // [n][k] (transposed), padded

    const int tid  = threadIdx.x;
    const int lane = tid & 31;
    const int wid  = tid >> 5;
    const int half = lane >> 4;
    const int lm   = lane & 15;
    const int wm   = wid & 3;           // 4 wave-rows
    const int wn   = wid >> 2;          // 2 wave-cols
    const int m0   = blockIdx.x * 128;
    const int n0   = blockIdx.y * 128;

    v8f acc[2][4] = {};

    for (int kb = 0; kb < Kdim; kb += 32) {
        // ---- stage A tile (128x32) ----
        {
            int row = tid >> 1, col = (tid & 1) * 16;
            const _Float16* gp = A + (size_t)(m0 + row) * Kdim + kb + col;
            *(uint4*)&As[row * 40 + col]     = *(const uint4*)gp;
            *(uint4*)&As[row * 40 + col + 8] = *(const uint4*)(gp + 8);
        }
        // ---- stage B tile transposed (32k x 128n -> Bt[n][k]) ----
        {
            int k = tid >> 3, n = (tid & 7) * 16;
            const _Float16* gp = Bw + (size_t)(kb + k) * Ndim + n0 + n;
            H16 t;
            t.u2[0] = *(const uint4*)gp;
            t.u2[1] = *(const uint4*)(gp + 8);
#pragma unroll
            for (int i = 0; i < 16; ++i) Bt[(n + i) * 40 + k] = t.h[i];
        }
        __syncthreads();

        // ---- fragments + WMMA ----
        v16h af[2], bf[4];
#pragma unroll
        for (int mt = 0; mt < 2; ++mt) {
            const _Float16* r = As + (wm * 32 + mt * 16 + lm) * 40;
            af[mt] = ld2x16(r + half * 8, r + 16 + half * 8);
        }
#pragma unroll
        for (int nt = 0; nt < 4; ++nt) {
            const _Float16* r = Bt + (wn * 64 + nt * 16 + lm) * 40;
            bf[nt] = ld2x16(r + half * 8, r + 16 + half * 8);
        }
#pragma unroll
        for (int mt = 0; mt < 2; ++mt)
#pragma unroll
            for (int nt = 0; nt < 4; ++nt)
                acc[mt][nt] = wmma32(af[mt], bf[nt], acc[mt][nt]);
        __syncthreads();
    }

    // ---- epilogue: bias, scale, store ----
#pragma unroll
    for (int mt = 0; mt < 2; ++mt)
#pragma unroll
        for (int nt = 0; nt < 4; ++nt) {
            int col   = n0 + wn * 64 + nt * 16 + lm;
            float bv  = bias ? bias[col] : 0.0f;
            int rbase = m0 + wm * 32 + mt * 16 + half * 8;
#pragma unroll
            for (int r = 0; r < 8; ++r) {
                float vv  = (acc[mt][nt][r] + bv) * scale;
                size_t ix = (size_t)(rbase + r) * Ndim + col;
                if (outh) outh[ix] = (_Float16)vv;
                if (outf) outf[ix] = vv;
            }
        }
}

// ---------------------------------------------------------------------------
// Attention: per (head, 32-query block).  Two-pass with full score rows
// resident in LDS (exploits CDNA5's 320 KB/WGP LDS):
//   phase 1:  S = Q * K^T  (scale folded into Q)          -> LDS fp32
//   phase 2:  row softmax in LDS (wave32 shfl reductions)
//   phase 3:  O = P * V  (V blocks transposed through LDS)
// Dynamic LDS: 256 KB scores + 4.5 KB Qstage + 17 KB Vt  ~= 278 KB.
// ---------------------------------------------------------------------------
#define ATTN_LDS_BYTES (32 * 2048 * 4 + 32 * 72 * 2 + 64 * 136 * 2)

__global__ void attn_kernel(const _Float16* __restrict__ Q,
                            const _Float16* __restrict__ K,
                            const _Float16* __restrict__ V,
                            _Float16* __restrict__ O) {
    extern __shared__ char smem[];
    float*    sc = (float*)smem;                                  // [32][2048]
    _Float16* Qs = (_Float16*)(smem + 32 * 2048 * 4);             // [32][72]
    _Float16* Vt = (_Float16*)(smem + 32 * 2048 * 4 + 32 * 72 * 2); // [64][136]

    const int tid  = threadIdx.x;
    const int lane = tid & 31;
    const int wid  = tid >> 5;
    const int half = lane >> 4;
    const int lm   = lane & 15;
    const int bh   = blockIdx.y;          // head index in [0, B*H)
    const int q0   = blockIdx.x * 32;     // query-block start

    // ---- stage Q block (32 x 64 f16) into LDS ----
    {
        int row = tid >> 3, col = (tid & 7) * 8;
        const _Float16* gp = Q + (size_t)(bh * SS + q0 + row) * HSZ + col;
        *(uint4*)&Qs[row * 72 + col] = *(const uint4*)gp;
    }
    __syncthreads();

    // ---- phase 1: scores = Q * K^T  (each wave: 16 of the 128 n-tiles) ----
    v16h af[2][2];
#pragma unroll
    for (int mt = 0; mt < 2; ++mt)
#pragma unroll
        for (int ks = 0; ks < 2; ++ks) {
            const _Float16* r = Qs + (mt * 16 + lm) * 72;
            int kb = ks * 32;
            af[mt][ks] = ld2x16(r + kb + half * 8, r + kb + 16 + half * 8);
        }

    for (int j = wid; j < 128; j += 8) {
        int n0 = j * 16;
        // B fragment = K^T: lane holds one key row, K contiguous along HS
        const _Float16* krow = K + (size_t)(bh * SS + n0 + lm) * HSZ;
        v8f a0 = {}, a1 = {};
#pragma unroll
        for (int ks = 0; ks < 2; ++ks) {
            int o = ks * 32 + half * 8;
            v16h bf = ld2x16(krow + o, krow + o + 16);
            a0 = wmma32(af[0][ks], bf, a0);
            a1 = wmma32(af[1][ks], bf, a1);
        }
        int col = n0 + lm;
#pragma unroll
        for (int r = 0; r < 8; ++r) {
            sc[(half * 8 + r) * 2048 + col]      = a0[r];
            sc[(16 + half * 8 + r) * 2048 + col] = a1[r];
        }
    }
    __syncthreads();

    // ---- phase 2: softmax over each of the 32 rows (8 threads / row) ----
    {
        int row = tid >> 3, sub = tid & 7;
        float* r = sc + row * 2048;
        int base = sub * 256;
        float mx = -3.0e38f;
        for (int i = 0; i < 256; ++i) mx = fmaxf(mx, r[base + i]);
        mx = fmaxf(mx, __shfl_xor(mx, 1));
        mx = fmaxf(mx, __shfl_xor(mx, 2));
        mx = fmaxf(mx, __shfl_xor(mx, 4));
        float sum = 0.0f;
        for (int i = 0; i < 256; ++i) {
            float e = __expf(r[base + i] - mx);
            r[base + i] = e;
            sum += e;
        }
        sum += __shfl_xor(sum, 1);
        sum += __shfl_xor(sum, 2);
        sum += __shfl_xor(sum, 4);
        float inv = 1.0f / sum;
        for (int i = 0; i < 256; ++i) r[base + i] *= inv;
    }

    // ---- phase 3: O = P * V.  Each wave owns one 16x16 output tile ----
    const int mt = wid >> 2, nt = wid & 3;
    v8f acc = {};
    const float* srow = sc + (mt * 16 + lm) * 2048;

    for (int kb = 0; kb < SS; kb += 128) {
        __syncthreads();   // previous Vt consumers done (also closes phase 2)
        {
            int kk = tid >> 1, nb = (tid & 1) * 32;
            const _Float16* vp = V + (size_t)(bh * SS + kb + kk) * HSZ + nb;
            H16 t0, t1;
            t0.u2[0] = *(const uint4*)vp;
            t0.u2[1] = *(const uint4*)(vp + 8);
            t1.u2[0] = *(const uint4*)(vp + 16);
            t1.u2[1] = *(const uint4*)(vp + 24);
#pragma unroll
            for (int i = 0; i < 16; ++i) {
                Vt[(nb + i) * 136 + kk]      = t0.h[i];
                Vt[(nb + 16 + i) * 136 + kk] = t1.h[i];
            }
        }
        __syncthreads();
#pragma unroll
        for (int k2 = 0; k2 < 128; k2 += 32) {
            H16 t;
            int o0 = kb + k2 + half * 8;
            int o1 = kb + k2 + 16 + half * 8;
#pragma unroll
            for (int i = 0; i < 8; ++i) {
                t.h[i]     = (_Float16)srow[o0 + i];
                t.h[8 + i] = (_Float16)srow[o1 + i];
            }
            const _Float16* vr = Vt + (nt * 16 + lm) * 136 + k2 + half * 8;
            v16h bf = ld2x16(vr, vr + 16);
            acc = wmma32(t.v, bf, acc);
        }
    }

    int col = nt * 16 + lm;
#pragma unroll
    for (int r = 0; r < 8; ++r) {
        int row = q0 + mt * 16 + half * 8 + r;
        O[(size_t)(bh * SS + row) * HSZ + col] = (_Float16)acc[r];
    }
}

// ---------------------------------------------------------------------------
// y = LayerNorm(x + proj) with std (population) + EPS, matching reference.
// One block per row (1024 cols, 256 threads x 4 elems).
// ---------------------------------------------------------------------------
__device__ __forceinline__ float block_sum(float v, float* red) {
#pragma unroll
    for (int m = 16; m >= 1; m >>= 1) v += __shfl_xor(v, m);
    int w = threadIdx.x >> 5, ln = threadIdx.x & 31;
    __syncthreads();
    if (ln == 0) red[w] = v;
    __syncthreads();
    float r = (ln < 8) ? red[ln] : 0.0f;
#pragma unroll
    for (int m = 4; m >= 1; m >>= 1) r += __shfl_xor(r, m);
    return __shfl(r, 0);
}

__global__ void resid_ln_kernel(const float* __restrict__ x,
                                const float* __restrict__ p,
                                const float* __restrict__ gamma,
                                const float* __restrict__ beta,
                                float* __restrict__ out) {
    __shared__ float red[8];
    const int row = blockIdx.x, t = threadIdx.x;
    const float* xr = x + (size_t)row * EE;
    const float* pr = p + (size_t)row * EE;

    float v[4], s = 0.0f;
#pragma unroll
    for (int j = 0; j < 4; ++j) {
        int c = t * 4 + j;
        v[j] = xr[c] + pr[c];
        s += v[j];
    }
    float mean = block_sum(s, red) * (1.0f / EE);

    float s2 = 0.0f;
#pragma unroll
    for (int j = 0; j < 4; ++j) {
        float d = v[j] - mean;
        s2 += d * d;
    }
    float var = block_sum(s2, red) * (1.0f / EE);
    float inv = 1.0f / (sqrtf(var) + 1e-8f);

#pragma unroll
    for (int j = 0; j < 4; ++j) {
        int c = t * 4 + j;
        out[(size_t)row * EE + c] = gamma[c] * ((v[j] - mean) * inv) + beta[c];
    }
}

// ---------------------------------------------------------------------------
// Launch: cvt -> QKV gemms -> attention -> O gemm -> residual+LN
// Workspace layout (bytes):
//   xh 0..8M | Wq 8M | Wk 10M | Wv 12M | Wo 14M | Qh 16M | Kh 24M | Vh 32M
//   attn_out 40M | proj(f32) 48M..64M
// ---------------------------------------------------------------------------
extern "C" void kernel_launch(void* const* d_in, const int* in_sizes, int n_in,
                              void* d_out, int out_size, void* d_ws, size_t ws_size,
                              hipStream_t stream) {
    const float* x     = (const float*)d_in[0];
    const float* Wq    = (const float*)d_in[1];
    const float* bq    = (const float*)d_in[2];
    const float* Wk    = (const float*)d_in[3];
    const float* bk    = (const float*)d_in[4];
    const float* Wv    = (const float*)d_in[5];
    const float* bv    = (const float*)d_in[6];
    const float* Wo    = (const float*)d_in[7];
    const float* bo    = (const float*)d_in[8];
    const float* gamma = (const float*)d_in[9];
    const float* beta  = (const float*)d_in[10];

    char* ws = (char*)d_ws;
    const size_t MB = 1u << 20;
    _Float16* xh  = (_Float16*)(ws + 0 * MB);
    _Float16* wqh = (_Float16*)(ws + 8 * MB);
    _Float16* wkh = (_Float16*)(ws + 10 * MB);
    _Float16* wvh = (_Float16*)(ws + 12 * MB);
    _Float16* woh = (_Float16*)(ws + 14 * MB);
    _Float16* Qh  = (_Float16*)(ws + 16 * MB);
    _Float16* Kh  = (_Float16*)(ws + 24 * MB);
    _Float16* Vh  = (_Float16*)(ws + 32 * MB);
    _Float16* Ah  = (_Float16*)(ws + 40 * MB);
    float*    Pf  = (float*)   (ws + 48 * MB);

    // 1) conversions
    cvt_f32_to_f16<<<(MM * EE) / 1024, 256, 0, stream>>>(x, xh, MM * EE);
    cvt_f32_to_f16<<<(EE * EE) / 1024, 256, 0, stream>>>(Wq, wqh, EE * EE);
    cvt_f32_to_f16<<<(EE * EE) / 1024, 256, 0, stream>>>(Wk, wkh, EE * EE);
    cvt_f32_to_f16<<<(EE * EE) / 1024, 256, 0, stream>>>(Wv, wvh, EE * EE);
    cvt_f32_to_f16<<<(EE * EE) / 1024, 256, 0, stream>>>(Wo, woh, EE * EE);

    // 2) QKV projections (1/sqrt(HS)=0.125 folded into Q)
    dim3 ggrid(MM / 128, EE / 128);
    gemm_f16_kernel<<<ggrid, 256, 0, stream>>>(xh, wqh, bq, Qh, nullptr,
                                               MM, EE, EE, 0.125f);
    gemm_f16_kernel<<<ggrid, 256, 0, stream>>>(xh, wkh, bk, Kh, nullptr,
                                               MM, EE, EE, 1.0f);
    gemm_f16_kernel<<<ggrid, 256, 0, stream>>>(xh, wvh, bv, Vh, nullptr,
                                               MM, EE, EE, 1.0f);

    // 3) attention (grid: 64 query blocks x 32 heads)
    attn_kernel<<<dim3(SS / 32, BB * HH), 256, ATTN_LDS_BYTES, stream>>>(
        Qh, Kh, Vh, Ah);

    // 4) output projection -> fp32
    gemm_f16_kernel<<<ggrid, 256, 0, stream>>>(Ah, woh, bo, nullptr, Pf,
                                               MM, EE, EE, 1.0f);

    // 5) residual + LayerNorm -> d_out (fp32)
    resid_ln_kernel<<<MM, 256, 0, stream>>>(x, Pf, gamma, beta, (float*)d_out);
}
